// RNNTDecoder_52149492908816
// MI455X (gfx1250) — matrix operand order repaired and checked by reading, not compiled, not run on hardware
//
#include <hip/hip_runtime.h>

typedef __attribute__((ext_vector_type(16))) _Float16 v16h;
typedef __attribute__((ext_vector_type(8)))  float    v8f;

#define B_TOT 32768
#define H     320
#define NCLS  34
#define NG    1280      // 4H gate columns (N' gate-interleaved)
#define KG    640       // 2H concat K
#define BM    64        // batch rows per block (4 M-tiles)
#define GK_TILES 20     // 640/32
#define GN_TILES 80     // 1280/16
#define PK_TILES 10     // 320/32
#define PN_TILES 20     // 320/16
#define JK_TILES 10
#define JN_TILES 3      // 48 padded class cols

#define TILE_HALFS 512  // 32 lanes * 16 halves per 16x32 / 32x16 f16 tile
#define WG_HALFS (GN_TILES*GK_TILES*TILE_HALFS)  // 819200
#define WP_HALFS (PN_TILES*PK_TILES*TILE_HALFS)  // 102400
#define WJ_HALFS (JN_TILES*JK_TILES*TILE_HALFS)  // 15360
#define W_TOTAL  (WG_HALFS+WP_HALFS+WJ_HALFS)    // 936960 halves (~1.9MB)
#define CONV_TOTAL (W_TOTAL + NG)                // + 1280 pre-summed gate biases (f32)

__device__ __forceinline__ float sigf(float x)  { return 1.0f / (1.0f + __expf(-x)); }
__device__ __forceinline__ float tanhf2(float x){ return 2.0f / (1.0f + __expf(-2.0f * x)) - 1.0f; }

// ---------------------------------------------------------------------------
// Prep: fp32 weights -> f16, pre-swizzled into WMMA B-fragment order.
// B tile (32x16 KxN), 16-bit: lane l<16: N=l, halves 0..7 = K 0..7, 8..15 = K 16..23;
// lane l>=16: N=l-16, halves 0..7 = K 8..15, 8..15 = K 24..31.  idx_in_tile = lane*16 + h.
// Gates region uses gate-INTERLEAVED columns: N' = 4*j + gate  (gate: i,f,g,o)
// so one 16-wide N-tile = 4 hidden units with their 4 gates in adjacent lanes.
// Also emits bsum[N'] = b_ih[n] + b_hh[n] (f32) after the halves.
// ---------------------------------------------------------------------------
__global__ __launch_bounds__(256)
void convert_weights(const float* __restrict__ Wih, const float* __restrict__ Whh,
                     const float* __restrict__ predW, const float* __restrict__ jointW,
                     const float* __restrict__ b_ih, const float* __restrict__ b_hh,
                     _Float16* __restrict__ wsH)
{
    int idx = blockIdx.x * 256 + threadIdx.x;
    if (idx >= CONV_TOTAL) return;

    if (idx >= W_TOTAL) {                         // pre-summed gate biases in N' order
        int np = idx - W_TOTAL;                   // 0..1279
        int n  = (np & 3) * H + (np >> 2);        // original 4H index
        float* bsum = (float*)(wsH + W_TOTAL);
        bsum[np] = b_ih[n] + b_hh[n];
        return;
    }

    int region, t, r, kTiles;
    if (idx < WG_HALFS)                 { region = 0; t = idx >> 9; r = idx & 511; kTiles = GK_TILES; }
    else if (idx < WG_HALFS + WP_HALFS) { region = 1; int i2 = idx - WG_HALFS;            t = i2 >> 9; r = i2 & 511; kTiles = PK_TILES; }
    else                                { region = 2; int i3 = idx - WG_HALFS - WP_HALFS; t = i3 >> 9; r = i3 & 511; kTiles = JK_TILES; }

    int n_tile = t / kTiles, k_tile = t - n_tile * kTiles;
    int lane = r >> 4, hh = r & 15;
    int n_local = lane & 15;
    int kk = (hh & 7) + 8 * (lane >> 4) + 16 * (hh >> 3);
    int np = n_tile * 16 + n_local;
    int k  = k_tile * 32 + kk;

    float v;
    if (region == 0) {
        int n = (np & 3) * H + (np >> 2);         // de-interleave: gate-major original row
        v = (k < H) ? Wih[n * H + k] : Whh[n * H + (k - H)];
    } else if (region == 1) {
        v = predW[np * H + k];
    } else {
        v = (np < NCLS) ? jointW[np * H + k] : 0.0f;
    }
    wsH[idx] = (_Float16)v;
}

// ---------------------------------------------------------------------------
// Fused RNN-T step: embed -> LSTM cell (in-register via quad ds_swizzle)
//                   -> pred proj -> joint -> argmax.
// One block = 64 batch rows (4 M-tiles), 8 wave32s; each B-fragment feeds
// 4 WMMAs (4x less L2 weight traffic than M=16 tiling).
// A-frag layout (16-bit 16x32): lane = m + 16*((kk>>3)&1), h = (kk&7)+8*(kk>>4).
// ---------------------------------------------------------------------------
__global__ __launch_bounds__(256)
void rnnt_fused(const int* __restrict__ x,
                const float* __restrict__ enc,
                const float* __restrict__ hin,
                const float* __restrict__ cin,
                const float* __restrict__ embW,
                const float* __restrict__ pred_b,
                const float* __restrict__ joint_b,
                const _Float16* __restrict__ wH,
                const float* __restrict__ bsum,
                float* __restrict__ out)
{
    extern __shared__ char smem[];
    _Float16* aBuf = (_Float16*)smem;              // 4 mt * 20 kt * 512 halves = 81920 B
    _Float16* hBuf = (_Float16*)(smem + 81920);    // 4 mt * 10 kt * 512 halves = 40960 B
    _Float16* zBuf = (_Float16*)smem;              // reuse aBuf region (40960 B)
    float*    jbuf = (float*)(smem + 81920);       // reuse hBuf region (64*48*4 = 12288 B)

    const int tid = threadIdx.x;
    const int wv  = tid >> 5;
    const int ln  = tid & 31;
    const int base = blockIdx.x * BM;

    // ---- stage 0: gather embedding + h state -> f16 A-frags ----
    for (int i = tid; i < BM * KG; i += 256) {
        int r = i / KG, k = i - r * KG;
        int b = base + r;
        float v;
        if (k < H) { int tok = x[b]; v = embW[tok * H + k]; }   // padding row is zeros
        else       { v = hin[(size_t)b * H + (k - H)]; }
        int mt = r >> 4, m = r & 15;
        int kt = k >> 5, kk = k & 31;
        int lane = m + 16 * ((kk >> 3) & 1);
        int hh = (kk & 7) + 8 * (kk >> 4);
        aBuf[(mt * GK_TILES + kt) * TILE_HALFS + lane * 16 + hh] = (_Float16)v;
    }
    __syncthreads();

    // ---- stage 1: gates GEMM [64,640]x[640,1280] + fused in-register LSTM ----
    {
        const _Float16* wg = wH;
        const int rhalf = 8 * (ln >> 4);           // row offset from C/D lane layout
        for (int ntl = 0; ntl < 10; ++ntl) {
            int nt = wv * 10 + ntl;
            v8f acc[4] = {};
            const _Float16* wtile = wg + (size_t)(nt * GK_TILES) * TILE_HALFS + ln * 16;
            // WGP-scope speculative prefetch of next N-tile's weight column
            // (locality 3 -> near-cache fill; harmless past end of region)
            __builtin_prefetch((const void*)(wtile + GK_TILES * TILE_HALFS), 0, 3);
            for (int kt = 0; kt < GK_TILES; ++kt) {
                v16h bm = *(const v16h*)(wtile + kt * TILE_HALFS);
                #pragma unroll
                for (int mt = 0; mt < 4; ++mt) {
                    v16h a = *(const v16h*)(aBuf + (mt * GK_TILES + kt) * TILE_HALFS + ln * 16);
                    acc[mt] = __builtin_amdgcn_wmma_f32_16x16x32_f16(false, a, false, bm,
                                                                     (short)0, acc[mt], false, false);
                }
            }
            // epilogue: bias add, quad-broadcast i/f/g/o across lanes, LSTM cell
            int colp = nt * 16 + (ln & 15);        // N' column: 4*j + gate
            int j = colp >> 2;
            float bsv = bsum[colp];
            int kt2 = j >> 5, kk = j & 31;
            int laneAdj = 16 * ((kk >> 3) & 1);
            int hh = (kk & 7) + 8 * (kk >> 4);
            #pragma unroll
            for (int mt = 0; mt < 4; ++mt) {
                #pragma unroll
                for (int r8 = 0; r8 < 8; ++r8) {
                    float gv = acc[mt][r8] + bsv;
                    int gi = __float_as_int(gv);
                    // group-of-4 full-share swizzles: broadcast lane (quad base + g)
                    float iv = __int_as_float(__builtin_amdgcn_ds_swizzle(gi, 0x8000));
                    float fv = __int_as_float(__builtin_amdgcn_ds_swizzle(gi, 0x8055));
                    float gg = __int_as_float(__builtin_amdgcn_ds_swizzle(gi, 0x80AA));
                    float ov = __int_as_float(__builtin_amdgcn_ds_swizzle(gi, 0x80FF));
                    if ((ln & 3) == 0) {           // one lane per hidden unit finishes it
                        int row = mt * 16 + r8 + rhalf;
                        int b = base + row;
                        float cn = sigf(fv) * cin[(size_t)b * H + j] + sigf(iv) * tanhf2(gg);
                        float hn = sigf(ov) * tanhf2(cn);
                        out[B_TOT + (size_t)b * H + j]                     = hn;  // h_new
                        out[B_TOT + (size_t)B_TOT * H + (size_t)b * H + j] = cn;  // c_new
                        hBuf[(mt * PK_TILES + kt2) * TILE_HALFS
                             + ((row & 15) + laneAdj) * 16 + hh] = (_Float16)hn;
                    }
                }
            }
        }
    }
    __syncthreads();

    // ---- stage 2: pred GEMM [64,320]x[320,320] + enc add -> z frags ----
    {
        const _Float16* wp = wH + WG_HALFS;
        const int rhalf = 8 * (ln >> 4);
        for (int nt = wv; nt < PN_TILES; nt += 8) {
            v8f acc[4] = {};
            const _Float16* wtile = wp + (size_t)(nt * PK_TILES) * TILE_HALFS + ln * 16;
            for (int kt = 0; kt < PK_TILES; ++kt) {
                v16h bm = *(const v16h*)(wtile + kt * TILE_HALFS);
                #pragma unroll
                for (int mt = 0; mt < 4; ++mt) {
                    v16h a = *(const v16h*)(hBuf + (mt * PK_TILES + kt) * TILE_HALFS + ln * 16);
                    acc[mt] = __builtin_amdgcn_wmma_f32_16x16x32_f16(false, a, false, bm,
                                                                     (short)0, acc[mt], false, false);
                }
            }
            int col = nt * 16 + (ln & 15);
            float pb = pred_b[col];
            int kt2 = col >> 5, kk = col & 31;
            int laneAdj = 16 * ((kk >> 3) & 1);
            int hh = (kk & 7) + 8 * (kk >> 4);
            #pragma unroll
            for (int mt = 0; mt < 4; ++mt) {
                #pragma unroll
                for (int r8 = 0; r8 < 8; ++r8) {
                    int row = mt * 16 + r8 + rhalf;
                    int b = base + row;
                    float z = acc[mt][r8] + enc[(size_t)b * H + col] + pb;
                    zBuf[(mt * PK_TILES + kt2) * TILE_HALFS
                         + ((row & 15) + laneAdj) * 16 + hh] = (_Float16)z;
                }
            }
        }
    }
    __syncthreads();

    // ---- stage 3: joint GEMM [64,320]x[320,48] (waves 0..2, 4 M-tiles each) ----
    if (wv < JN_TILES) {
        const _Float16* wj = wH + WG_HALFS + WP_HALFS;
        const int rhalf = 8 * (ln >> 4);
        int nt = wv;
        v8f acc[4] = {};
        const _Float16* wtile = wj + (size_t)(nt * JK_TILES) * TILE_HALFS + ln * 16;
        for (int kt = 0; kt < JK_TILES; ++kt) {
            v16h bm = *(const v16h*)(wtile + kt * TILE_HALFS);
            #pragma unroll
            for (int mt = 0; mt < 4; ++mt) {
                v16h a = *(const v16h*)(zBuf + (mt * PK_TILES + kt) * TILE_HALFS + ln * 16);
                acc[mt] = __builtin_amdgcn_wmma_f32_16x16x32_f16(false, a, false, bm,
                                                                 (short)0, acc[mt], false, false);
            }
        }
        int col = nt * 16 + (ln & 15);
        #pragma unroll
        for (int mt = 0; mt < 4; ++mt)
            #pragma unroll
            for (int r8 = 0; r8 < 8; ++r8)
                jbuf[(mt * 16 + r8 + rhalf) * 48 + col] = acc[mt][r8];
    }
    __syncthreads();

    // ---- stage 4: per-row argmax over 34 classes ----
    if (tid < BM) {
        int b = base + tid;
        float best = -3.4e38f;
        int bi = 0;
        for (int v = 0; v < NCLS; ++v) {
            float val = jbuf[tid * 48 + v] + joint_b[v];
            if (val > best) { best = val; bi = v; }
        }
        out[b] = (float)bi;   // dec stored as float
    }
}

extern "C" void kernel_launch(void* const* d_in, const int* in_sizes, int n_in,
                              void* d_out, int out_size, void* d_ws, size_t ws_size,
                              hipStream_t stream) {
    const int*   x      = (const int*)  d_in[0];
    const float* enc    = (const float*)d_in[1];
    const float* h      = (const float*)d_in[2];
    const float* c      = (const float*)d_in[3];
    const float* embW   = (const float*)d_in[4];
    const float* Wih    = (const float*)d_in[5];
    const float* Whh    = (const float*)d_in[6];
    const float* bih    = (const float*)d_in[7];
    const float* bhh    = (const float*)d_in[8];
    const float* predW  = (const float*)d_in[9];
    const float* predb  = (const float*)d_in[10];
    const float* jointW = (const float*)d_in[11];
    const float* jointb = (const float*)d_in[12];
    _Float16* wsH = (_Float16*)d_ws;
    const float* bsum = (const float*)(wsH + W_TOTAL);
    float* out = (float*)d_out;

    // 1) f16 weight pack (gate-interleaved) + pre-summed gate biases (~1.9 MB, L2-resident)
    convert_weights<<<(CONV_TOTAL + 255) / 256, 256, 0, stream>>>(Wih, Whh, predW, jointW,
                                                                  bih, bhh, wsH);

    // 2) fused decoder step: 512 blocks x 256 threads, 120 KB dynamic LDS (2 blocks/WGP)
    rnnt_fused<<<B_TOT / BM, 256, 122880, stream>>>(x, enc, h, c, embW,
                                                    predb, jointb, wsH, bsum, out);
}
